// EncodeProcessDecode_5497558139544
// MI455X (gfx1250) — matrix operand order repaired and checked by reading, not compiled, not run on hardware
//
#include <hip/hip_runtime.h>
#include <cmath>
#include <stdint.h>

typedef _Float16 half_t;
typedef _Float16 v16h __attribute__((ext_vector_type(16)));
typedef _Float16 v8h  __attribute__((ext_vector_type(8)));
typedef float    v8f  __attribute__((ext_vector_type(8)));

#define NNODE 20000
#define NEDGE 100000
#define LATD  128
#define NSTEP 6

// ---- CDNA5 async global->LDS copy (ASYNCcnt-tracked), with safe fallback ----
#if defined(__AMDGCN__) && \
    __has_builtin(__builtin_amdgcn_global_load_async_to_lds_b128) && \
    __has_builtin(__builtin_amdgcn_s_wait_asynccnt)
#define HAVE_ASYNC_LDS 1
#endif

__device__ __forceinline__ void cp16(const half_t* g, half_t* l) {
#ifdef HAVE_ASYNC_LDS
    // builtin signature (from diagnostic): (v4i AS1*, v4i AS3*, imm int, imm int)
    typedef int v4ig __attribute__((vector_size(16)));
    typedef v4ig __attribute__((address_space(1)))* gp_t;
    typedef v4ig __attribute__((address_space(3)))* lp_t;
    __builtin_amdgcn_global_load_async_to_lds_b128(
        (gp_t)(uintptr_t)g,       // global VA passes through unchanged
        (lp_t)(uintptr_t)l,       // low 32 bits of generic ptr == LDS offset
        0, 0);
#else
    *(float4*)l = *(const float4*)g;
#endif
}
__device__ __forceinline__ void async_join() {
#ifdef HAVE_ASYNC_LDS
    __builtin_amdgcn_s_wait_asynccnt(0);
#endif
}

// ---------------- WMMA fragment helpers (ISA 7.12.2 f16 layouts) ----------------
__device__ __forceinline__ v16h a_frag(const half_t* lds, int ld, int m, int k0, int h) {
    v8h lo = *(const v8h*)(lds + m * ld + k0 + h * 8);
    v8h hi = *(const v8h*)(lds + m * ld + k0 + 16 + h * 8);
    v16h a;
#pragma unroll
    for (int i = 0; i < 8; ++i) { a[i] = lo[i]; a[i + 8] = hi[i]; }
    return a;
}
__device__ __forceinline__ v16h b_frag(const half_t* Wt, int K, int n, int k0, int h) {
    v8h lo = *(const v8h*)(Wt + (size_t)n * K + k0 + h * 8);
    v8h hi = *(const v8h*)(Wt + (size_t)n * K + k0 + 16 + h * 8);
    v16h b;
#pragma unroll
    for (int i = 0; i < 8; ++i) { b[i] = lo[i]; b[i + 8] = hi[i]; }
    return b;
}

// ---------------- weight convert + transpose: dst[n*K+k] = (f16)src[k*128+n] ----
__global__ void k_wcvt(const float* __restrict__ src, half_t* __restrict__ dst, int K) {
    int idx = blockIdx.x * 256 + threadIdx.x;
    if (idx >= K * 128) return;
    int n = idx / K, k = idx % K;
    dst[idx] = (half_t)src[k * 128 + n];
}

__global__ void k_zero(float* __restrict__ p, int n) {
    int i = blockIdx.x * 256 + threadIdx.x;
    if (i < n) p[i] = 0.f;
}

// ---------------- fused encoder: norm -> L1(VALU) -> ReLU -> L2(WMMA) -> ReLU -> LN
__global__ __launch_bounds__(256) void k_encode(
    const float* __restrict__ x, const float* __restrict__ mean, const float* __restrict__ stdv,
    int in_feat,
    const float* __restrict__ W1, const float* __restrict__ b1,
    const half_t* __restrict__ W2t, const float* __restrict__ b2,
    const float* __restrict__ g, const float* __restrict__ beta,
    float* __restrict__ lat, half_t* __restrict__ latH)
{
    __shared__ float  xn[16 * 16];
    __shared__ half_t ldsH1[16 * LATD];
    __shared__ float  ldsC[16 * LATD];
    const int t = threadIdx.x;
    const int r0 = blockIdx.x * 16;

    for (int idx = t; idx < 16 * in_feat; idx += 256) {
        int row = idx / in_feat, k = idx % in_feat;
        xn[row * 16 + k] = (x[(size_t)(r0 + row) * in_feat + k] - mean[k]) / stdv[k];
    }
    __syncthreads();

    { // layer 1 (K = in_feat, tiny) in VALU
        int row = t >> 4, c0 = (t & 15) * 8;
        for (int i = 0; i < 8; ++i) {
            int c = c0 + i;
            float acc = b1[c];
            for (int k = 0; k < in_feat; ++k) acc += xn[row * 16 + k] * W1[k * LATD + c];
            ldsH1[row * LATD + c] = (half_t)(acc > 0.f ? acc : 0.f);
        }
    }
    __syncthreads();

    const int h = (t >> 4) & 1, nl = t & 15, n = ((t >> 5) * 16) + nl;
    v8f acc2 = {};
#pragma unroll
    for (int k0 = 0; k0 < LATD; k0 += 32) {
        v16h a = a_frag(ldsH1, LATD, nl, k0, h);
        v16h b = b_frag(W2t, LATD, n, k0, h);
        acc2 = __builtin_amdgcn_wmma_f32_16x16x32_f16(false, a, false, b, (short)0, acc2, false, false);
    }
    {
        float bb = b2[n];
#pragma unroll
        for (int r = 0; r < 8; ++r) {
            float v = acc2[r] + bb;
            ldsC[(h * 8 + r) * LATD + n] = v > 0.f ? v : 0.f;
        }
    }
    __syncthreads();

    { // LayerNorm per row (16 lanes/row), store f32 latent + f16 mirror
        const int row = t >> 4, sub = t & 15;
        float s = 0.f, ss = 0.f, vals[8];
#pragma unroll
        for (int i = 0; i < 8; ++i) { float v = ldsC[row * LATD + sub * 8 + i]; vals[i] = v; s += v; ss += v * v; }
#pragma unroll
        for (int m = 1; m < 16; m <<= 1) { s += __shfl_xor(s, m); ss += __shfl_xor(ss, m); }
        float mu = s * (1.f / LATD);
        float rstd = rsqrtf(ss * (1.f / LATD) - mu * mu + 1e-5f);
        size_t base = (size_t)(r0 + row) * LATD;
#pragma unroll
        for (int i = 0; i < 8; ++i) {
            int c = sub * 8 + i;
            float v = (vals[i] - mu) * rstd * g[c] + beta[c];
            lat[base + c] = v;
            latH[base + c] = (half_t)v;
        }
    }
}

// ---------------- fused edge block (32 rows): gather -> MLP(WMMA x2 tiles) -> LN -> residual+scatter
__global__ __launch_bounds__(256) void k_edge_block(
    const int* __restrict__ snd, const int* __restrict__ rcv,
    const half_t* __restrict__ nodeH, half_t* __restrict__ edgeH,
    float* __restrict__ edge_lat, float* __restrict__ aggr,
    const half_t* __restrict__ W1t, const float* __restrict__ b1,
    const half_t* __restrict__ W2t, const float* __restrict__ b2,
    const float* __restrict__ g, const float* __restrict__ beta)
{
    // ldsA (32x384 f16 = 24KB) aliased by ldsC (32x128 f32 = 16KB); ldsH1 separate (8KB)
    __shared__ __align__(16) char smem[32 * 384 * 2 + 32 * LATD * 2];
    half_t* ldsA  = (half_t*)smem;
    float*  ldsC  = (float*)smem;
    half_t* ldsH1 = (half_t*)(smem + 32 * 384 * 2);
    __shared__ int sS[32], sR[32];
    const int t = threadIdx.x;
    const int e0 = blockIdx.x * 32;

    if (t < 32) { sS[t] = snd[e0 + t]; sR[t] = rcv[e0 + t]; }
    __syncthreads();

    // stage A = [nodeH[s] | nodeH[r] | edgeH[e]] : 32 rows x 48 chunks of 16B (async)
#pragma unroll
    for (int j = 0; j < 6; ++j) {
        int c = t + j * 256;
        int row = c / 48, cc = c % 48, col0 = cc * 8;
        const half_t* src;
        if (col0 < 128)      src = nodeH + (size_t)sS[row] * LATD + col0;
        else if (col0 < 256) src = nodeH + (size_t)sR[row] * LATD + (col0 - 128);
        else                 src = edgeH + (size_t)(e0 + row) * LATD + (col0 - 256);
        cp16(src, ldsA + row * 384 + col0);
    }
    async_join();
    __syncthreads();

    const int h = (t >> 4) & 1, nl = t & 15, n = ((t >> 5) * 16) + nl;

    // ---- layer 1: K=384, two row-tiles share every B fragment ----
    v8f acc0 = {}, acc1 = {};
#pragma unroll
    for (int k0 = 0; k0 < 384; k0 += 32) {
        v16h a0 = a_frag(ldsA, 384, nl, k0, h);
        v16h a1 = a_frag(ldsA, 384, 16 + nl, k0, h);
        v16h b  = b_frag(W1t, 384, n, k0, h);
        acc0 = __builtin_amdgcn_wmma_f32_16x16x32_f16(false, a0, false, b, (short)0, acc0, false, false);
        acc1 = __builtin_amdgcn_wmma_f32_16x16x32_f16(false, a1, false, b, (short)0, acc1, false, false);
    }
    {
        float bb = b1[n];
#pragma unroll
        for (int r = 0; r < 8; ++r) {
            float v0 = acc0[r] + bb;
            float v1 = acc1[r] + bb;
            ldsH1[(h * 8 + r) * LATD + n]        = (half_t)(v0 > 0.f ? v0 : 0.f);
            ldsH1[(16 + h * 8 + r) * LATD + n]   = (half_t)(v1 > 0.f ? v1 : 0.f);
        }
    }
    __syncthreads();   // layer-1 done everywhere; ldsA region now dead -> reusable as ldsC

    // ---- layer 2: K=128 ----
    acc0 = (v8f){}; acc1 = (v8f){};
#pragma unroll
    for (int k0 = 0; k0 < LATD; k0 += 32) {
        v16h a0 = a_frag(ldsH1, LATD, nl, k0, h);
        v16h a1 = a_frag(ldsH1, LATD, 16 + nl, k0, h);
        v16h b  = b_frag(W2t, LATD, n, k0, h);
        acc0 = __builtin_amdgcn_wmma_f32_16x16x32_f16(false, a0, false, b, (short)0, acc0, false, false);
        acc1 = __builtin_amdgcn_wmma_f32_16x16x32_f16(false, a1, false, b, (short)0, acc1, false, false);
    }
    {
        float bb = b2[n];
#pragma unroll
        for (int r = 0; r < 8; ++r) {
            float v0 = acc0[r] + bb;
            float v1 = acc1[r] + bb;
            ldsC[(h * 8 + r) * LATD + n]      = v0 > 0.f ? v0 : 0.f;
            ldsC[(16 + h * 8 + r) * LATD + n] = v1 > 0.f ? v1 : 0.f;
        }
    }
    __syncthreads();

    { // LN (8 lanes per row), residual into edge latent, atomic scatter into aggr
        const int row = t >> 3, sub = t & 7;
        float s = 0.f, ss = 0.f, vals[16];
#pragma unroll
        for (int i = 0; i < 16; ++i) { float v = ldsC[row * LATD + sub * 16 + i]; vals[i] = v; s += v; ss += v * v; }
#pragma unroll
        for (int m = 1; m < 8; m <<= 1) { s += __shfl_xor(s, m); ss += __shfl_xor(ss, m); }
        float mu = s * (1.f / LATD);
        float rstd = rsqrtf(ss * (1.f / LATD) - mu * mu + 1e-5f);
        const int e = e0 + row, rv = sR[row];
        size_t ebase = (size_t)e * LATD, abase = (size_t)rv * LATD;
#pragma unroll
        for (int i = 0; i < 16; ++i) {
            int c = sub * 16 + i;
            float ne = (vals[i] - mu) * rstd * g[c] + beta[c];
            float upd = edge_lat[ebase + c] + ne;
            edge_lat[ebase + c] = upd;
            edgeH[ebase + c] = (half_t)upd;
            atomicAdd(&aggr[abase + c], ne);   // no-return -> global_atomic_add_f32
        }
    }
}

// ---------------- fused node block (32 rows): concat(nodeH, aggr) -> MLP -> LN -> residual
__global__ __launch_bounds__(256) void k_node_block(
    const half_t* __restrict__ nodeH_in, const float* __restrict__ aggr,
    float* __restrict__ node_lat, half_t* __restrict__ nodeH_out,
    const half_t* __restrict__ W1t, const float* __restrict__ b1,
    const half_t* __restrict__ W2t, const float* __restrict__ b2,
    const float* __restrict__ g, const float* __restrict__ beta)
{
    // ldsA (32x256 f16 = 16KB) aliased by ldsC (32x128 f32 = 16KB); ldsH1 separate (8KB)
    __shared__ __align__(16) char smem[32 * 256 * 2 + 32 * LATD * 2];
    half_t* ldsA  = (half_t*)smem;
    float*  ldsC  = (float*)smem;
    half_t* ldsH1 = (half_t*)(smem + 32 * 256 * 2);
    const int t = threadIdx.x;
    const int n0row = blockIdx.x * 32;

    // stage A = [nodeH | (f16)aggr] : 32 rows x 32 chunks of 16B
#pragma unroll
    for (int j = 0; j < 4; ++j) {
        int c = t + j * 256;
        int row = c / 32, cc = c % 32, col0 = cc * 8;
        if (col0 < 128) {
            cp16(nodeH_in + (size_t)(n0row + row) * LATD + col0, ldsA + row * 256 + col0);
        } else {
            const float* sp = aggr + (size_t)(n0row + row) * LATD + (col0 - 128);
            half_t tmp[8];
#pragma unroll
            for (int i = 0; i < 8; ++i) tmp[i] = (half_t)sp[i];
            *(float4*)(ldsA + row * 256 + col0) = *(const float4*)tmp;
        }
    }
    async_join();
    __syncthreads();

    const int h = (t >> 4) & 1, nl = t & 15, n = ((t >> 5) * 16) + nl;

    v8f acc0 = {}, acc1 = {};
#pragma unroll
    for (int k0 = 0; k0 < 256; k0 += 32) {
        v16h a0 = a_frag(ldsA, 256, nl, k0, h);
        v16h a1 = a_frag(ldsA, 256, 16 + nl, k0, h);
        v16h b  = b_frag(W1t, 256, n, k0, h);
        acc0 = __builtin_amdgcn_wmma_f32_16x16x32_f16(false, a0, false, b, (short)0, acc0, false, false);
        acc1 = __builtin_amdgcn_wmma_f32_16x16x32_f16(false, a1, false, b, (short)0, acc1, false, false);
    }
    {
        float bb = b1[n];
#pragma unroll
        for (int r = 0; r < 8; ++r) {
            float v0 = acc0[r] + bb;
            float v1 = acc1[r] + bb;
            ldsH1[(h * 8 + r) * LATD + n]      = (half_t)(v0 > 0.f ? v0 : 0.f);
            ldsH1[(16 + h * 8 + r) * LATD + n] = (half_t)(v1 > 0.f ? v1 : 0.f);
        }
    }
    __syncthreads();

    acc0 = (v8f){}; acc1 = (v8f){};
#pragma unroll
    for (int k0 = 0; k0 < LATD; k0 += 32) {
        v16h a0 = a_frag(ldsH1, LATD, nl, k0, h);
        v16h a1 = a_frag(ldsH1, LATD, 16 + nl, k0, h);
        v16h b  = b_frag(W2t, LATD, n, k0, h);
        acc0 = __builtin_amdgcn_wmma_f32_16x16x32_f16(false, a0, false, b, (short)0, acc0, false, false);
        acc1 = __builtin_amdgcn_wmma_f32_16x16x32_f16(false, a1, false, b, (short)0, acc1, false, false);
    }
    {
        float bb = b2[n];
#pragma unroll
        for (int r = 0; r < 8; ++r) {
            float v0 = acc0[r] + bb;
            float v1 = acc1[r] + bb;
            ldsC[(h * 8 + r) * LATD + n]      = v0 > 0.f ? v0 : 0.f;
            ldsC[(16 + h * 8 + r) * LATD + n] = v1 > 0.f ? v1 : 0.f;
        }
    }
    __syncthreads();

    {
        const int row = t >> 3, sub = t & 7;
        float s = 0.f, ss = 0.f, vals[16];
#pragma unroll
        for (int i = 0; i < 16; ++i) { float v = ldsC[row * LATD + sub * 16 + i]; vals[i] = v; s += v; ss += v * v; }
#pragma unroll
        for (int m = 1; m < 8; m <<= 1) { s += __shfl_xor(s, m); ss += __shfl_xor(ss, m); }
        float mu = s * (1.f / LATD);
        float rstd = rsqrtf(ss * (1.f / LATD) - mu * mu + 1e-5f);
        size_t base = (size_t)(n0row + row) * LATD;
#pragma unroll
        for (int i = 0; i < 16; ++i) {
            int c = sub * 16 + i;
            float nn = (vals[i] - mu) * rstd * g[c] + beta[c];
            float upd = node_lat[base + c] + nn;
            node_lat[base + c] = upd;
            nodeH_out[base + c] = (half_t)upd;
        }
    }
}

// ---------------- decoder: 128->8 swish 8->5, * dt ----------------
__global__ void k_decode(const float* __restrict__ lat,
                         const float* __restrict__ W1, const float* __restrict__ b1,
                         const float* __restrict__ W2, const float* __restrict__ b2,
                         float* __restrict__ out)
{
    int n = blockIdx.x * 256 + threadIdx.x;
    if (n >= NNODE) return;
    float hd[8];
#pragma unroll
    for (int j = 0; j < 8; ++j) hd[j] = b1[j];
    const float* lp = lat + (size_t)n * LATD;
    for (int k = 0; k < LATD; ++k) {
        float x = lp[k];
#pragma unroll
        for (int j = 0; j < 8; ++j) hd[j] += x * W1[k * 8 + j];
    }
#pragma unroll
    for (int j = 0; j < 8; ++j) hd[j] = hd[j] * (1.f / (1.f + __expf(-hd[j])));
#pragma unroll
    for (int tt = 0; tt < 5; ++tt) {
        float d = b2[tt];
#pragma unroll
        for (int j = 0; j < 8; ++j) d += hd[j] * W2[j * 5 + tt];
        out[(size_t)tt * NNODE + n] = d * (float)(tt + 1);
    }
}

// ------------------------------------------------------------------
extern "C" void kernel_launch(void* const* d_in, const int* in_sizes, int n_in,
                              void* d_out, int out_size, void* d_ws, size_t ws_size,
                              hipStream_t stream) {
    const float* node_features = (const float*)d_in[0];
    const float* edge_features = (const float*)d_in[1];
    const int*   senders       = (const int*)d_in[2];
    const int*   receivers     = (const int*)d_in[3];
    const float* node_mean = (const float*)d_in[4];
    const float* node_std  = (const float*)d_in[5];
    const float* edge_mean = (const float*)d_in[6];
    const float* edge_std  = (const float*)d_in[7];
    const float* enc_n_W1 = (const float*)d_in[8];
    const float* enc_n_b1 = (const float*)d_in[9];
    const float* enc_n_W2 = (const float*)d_in[10];
    const float* enc_n_b2 = (const float*)d_in[11];
    const float* enc_n_g    = (const float*)d_in[12];
    const float* enc_n_beta = (const float*)d_in[13];
    const float* enc_e_W1 = (const float*)d_in[14];
    const float* enc_e_b1 = (const float*)d_in[15];
    const float* enc_e_W2 = (const float*)d_in[16];
    const float* enc_e_b2 = (const float*)d_in[17];
    const float* enc_e_g    = (const float*)d_in[18];
    const float* enc_e_beta = (const float*)d_in[19];
    const float* blk_e_W1 = (const float*)d_in[20];
    const float* blk_e_b1 = (const float*)d_in[21];
    const float* blk_e_W2 = (const float*)d_in[22];
    const float* blk_e_b2 = (const float*)d_in[23];
    const float* blk_e_g    = (const float*)d_in[24];
    const float* blk_e_beta = (const float*)d_in[25];
    const float* blk_n_W1 = (const float*)d_in[26];
    const float* blk_n_b1 = (const float*)d_in[27];
    const float* blk_n_W2 = (const float*)d_in[28];
    const float* blk_n_b2 = (const float*)d_in[29];
    const float* blk_n_g    = (const float*)d_in[30];
    const float* blk_n_beta = (const float*)d_in[31];
    const float* dec_W1 = (const float*)d_in[32];
    const float* dec_b1 = (const float*)d_in[33];
    const float* dec_W2 = (const float*)d_in[34];
    const float* dec_b2 = (const float*)d_in[35];

    // ---- workspace carve-up (256B aligned) ----
    char* wp = (char*)d_ws;
    auto carve = [&](size_t bytes) -> void* {
        void* p = (void*)wp;
        wp += (bytes + 255) & ~(size_t)255;
        return p;
    };
    float*  node_lat = (float*)carve((size_t)NNODE * LATD * 4);
    float*  edge_lat = (float*)carve((size_t)NEDGE * LATD * 4);
    float*  aggr     = (float*)carve((size_t)NNODE * LATD * 4);
    half_t* nodeH    = (half_t*)carve((size_t)NNODE * LATD * 2);
    half_t* edgeH    = (half_t*)carve((size_t)NEDGE * LATD * 2);
    half_t* encnW2t  = (half_t*)carve((size_t)128 * 128 * 2);
    half_t* enceW2t  = (half_t*)carve((size_t)128 * 128 * 2);
    half_t* blkeW1t  = (half_t*)carve((size_t)NSTEP * 384 * 128 * 2);
    half_t* blkeW2t  = (half_t*)carve((size_t)NSTEP * 128 * 128 * 2);
    half_t* blknW1t  = (half_t*)carve((size_t)NSTEP * 256 * 128 * 2);
    half_t* blknW2t  = (half_t*)carve((size_t)NSTEP * 128 * 128 * 2);

    // ---- convert + transpose weights to f16 Wt[n][k] ----
    auto cvt = [&](const float* src, half_t* dst, int K) {
        int total = K * 128;
        k_wcvt<<<(total + 255) / 256, 256, 0, stream>>>(src, dst, K);
    };
    cvt(enc_n_W2, encnW2t, 128);
    cvt(enc_e_W2, enceW2t, 128);
    for (int s = 0; s < NSTEP; ++s) {
        cvt(blk_e_W1 + (size_t)s * 384 * 128, blkeW1t + (size_t)s * 384 * 128, 384);
        cvt(blk_e_W2 + (size_t)s * 128 * 128, blkeW2t + (size_t)s * 128 * 128, 128);
        cvt(blk_n_W1 + (size_t)s * 256 * 128, blknW1t + (size_t)s * 256 * 128, 256);
        cvt(blk_n_W2 + (size_t)s * 128 * 128, blknW2t + (size_t)s * 128 * 128, 128);
    }

    // ---- encode ----
    k_encode<<<NNODE / 16, 256, 0, stream>>>(
        node_features, node_mean, node_std, 11,
        enc_n_W1, enc_n_b1, encnW2t, enc_n_b2, enc_n_g, enc_n_beta,
        node_lat, nodeH);
    k_encode<<<NEDGE / 16, 256, 0, stream>>>(
        edge_features, edge_mean, edge_std, 5,
        enc_e_W1, enc_e_b1, enceW2t, enc_e_b2, enc_e_g, enc_e_beta,
        edge_lat, edgeH);

    // ---- process: 6 message-passing steps ----
    for (int s = 0; s < NSTEP; ++s) {
        k_zero<<<(NNODE * LATD + 255) / 256, 256, 0, stream>>>(aggr, NNODE * LATD);
        k_edge_block<<<NEDGE / 32, 256, 0, stream>>>(
            senders, receivers, nodeH, edgeH, edge_lat, aggr,
            blkeW1t + (size_t)s * 384 * 128, blk_e_b1 + s * 128,
            blkeW2t + (size_t)s * 128 * 128, blk_e_b2 + s * 128,
            blk_e_g + s * 128, blk_e_beta + s * 128);
        k_node_block<<<NNODE / 32, 256, 0, stream>>>(
            nodeH, aggr, node_lat, nodeH,
            blknW1t + (size_t)s * 256 * 128, blk_n_b1 + s * 128,
            blknW2t + (size_t)s * 128 * 128, blk_n_b2 + s * 128,
            blk_n_g + s * 128, blk_n_beta + s * 128);
    }

    // ---- decode ----
    k_decode<<<(NNODE + 255) / 256, 256, 0, stream>>>(
        node_lat, dec_W1, dec_b1, dec_W2, dec_b2, (float*)d_out);
}
